// TextMambaBlock_62964220559381
// MI455X (gfx1250) — matrix operand order repaired and checked by reading, not compiled, not run on hardware
//
#include <hip/hip_runtime.h>
#include <hip/hip_bf16.h>

#define D_MODEL 1024
#define INNER   2048
#define NPROJ   (2*INNER)
#define BATCH   8
#define SEQ     1024
#define ROWS    (BATCH*SEQ)   /* 8192 token rows */
#define LN_EPS  1e-5f

// GEMM tiling
#define BM   128
#define BN   64
#define BKK  32
#define ASTR (BKK + 8)        /* 40 elems = 80B row stride: conflict-free */

typedef __attribute__((ext_vector_type(16))) __bf16 bf16x16;
typedef __attribute__((ext_vector_type(8)))  __bf16 bf16x8;
typedef __attribute__((ext_vector_type(8)))  float  f32x8;
typedef __attribute__((ext_vector_type(4)))  int    i32x4;

#define AS1 __attribute__((address_space(1)))
#define AS3 __attribute__((address_space(3)))

#if defined(__HIP_DEVICE_COMPILE__) && \
    __has_builtin(__builtin_amdgcn_global_load_async_to_lds_b128) && \
    __has_builtin(__builtin_amdgcn_s_wait_asynccnt)
#define HAVE_ASYNC 1
#else
#define HAVE_ASYNC 0
#endif

__device__ __forceinline__ __bf16 f2bf(float f) {
    unsigned u = __float_as_uint(f);
    u += 0x7FFFu + ((u >> 16) & 1u);          // round-to-nearest-even
    unsigned short h = (unsigned short)(u >> 16);
    return __builtin_bit_cast(__bf16, h);
}

__device__ __forceinline__ float sigmf(float x) {
    return 1.0f / (1.0f + __expf(-x));
}

// 16B global -> LDS copy: async on CDNA5 (ASYNCcnt), sync fallback otherwise.
__device__ __forceinline__ void cp16_g2l(const __bf16* g, __bf16* l) {
#if HAVE_ASYNC
    __builtin_amdgcn_global_load_async_to_lds_b128(
        (AS1 i32x4*)g, (AS3 i32x4*)l, 0, 0);
#else
    *(bf16x8*)l = *(const bf16x8*)g;
#endif
}
__device__ __forceinline__ void wait_async_le3() {
#if HAVE_ASYNC
    __builtin_amdgcn_s_wait_asynccnt(3);
#endif
}
__device__ __forceinline__ void wait_async_zero() {
#if HAVE_ASYNC
    __builtin_amdgcn_s_wait_asynccnt(0);
#endif
}

__device__ __forceinline__ bf16x16 frag16(const __bf16* lo, const __bf16* hi) {
    bf16x8 l = *(const bf16x8*)lo;
    bf16x8 h = *(const bf16x8*)hi;
    bf16x16 r;
    #pragma unroll
    for (int i = 0; i < 8; ++i) { r[i] = l[i]; r[8 + i] = h[i]; }
    return r;
}

// ---------------------------------------------------------------- LayerNorm
__global__ void ln_kernel(const float* __restrict__ x,
                          const float* __restrict__ gamma,
                          const float* __restrict__ beta,
                          __bf16* __restrict__ h) {
    __shared__ float red[256];
    const int row = blockIdx.x;
    const float* xr = x + (size_t)row * D_MODEL;
    float v[4]; float s = 0.f;
    #pragma unroll
    for (int i = 0; i < 4; ++i) { v[i] = xr[threadIdx.x + 256 * i]; s += v[i]; }
    red[threadIdx.x] = s; __syncthreads();
    for (int off = 128; off > 0; off >>= 1) {
        if (threadIdx.x < off) red[threadIdx.x] += red[threadIdx.x + off];
        __syncthreads();
    }
    const float mu = red[0] * (1.0f / D_MODEL);
    __syncthreads();
    float s2 = 0.f;
    #pragma unroll
    for (int i = 0; i < 4; ++i) { float d = v[i] - mu; s2 += d * d; }
    red[threadIdx.x] = s2; __syncthreads();
    for (int off = 128; off > 0; off >>= 1) {
        if (threadIdx.x < off) red[threadIdx.x] += red[threadIdx.x + off];
        __syncthreads();
    }
    const float inv = rsqrtf(red[0] * (1.0f / D_MODEL) + LN_EPS);
    __bf16* hr = h + (size_t)row * D_MODEL;
    #pragma unroll
    for (int i = 0; i < 4; ++i) {
        int c = threadIdx.x + 256 * i;
        hr[c] = f2bf((v[i] - mu) * inv * gamma[c] + beta[c]);
    }
}

// ------------------------------------------- weight fp32[K][N] -> bf16[N][K]
__global__ void cvt_t_kernel(const float* __restrict__ W, __bf16* __restrict__ Wt,
                             int K, int N) {
    size_t total = (size_t)K * N;
    for (size_t e = (size_t)blockIdx.x * blockDim.x + threadIdx.x; e < total;
         e += (size_t)gridDim.x * blockDim.x) {
        size_t n = e / (size_t)K, k = e % (size_t)K;
        Wt[e] = f2bf(W[k * (size_t)N + n]);
    }
}

// ---------------------------------------------------------------- WMMA GEMM
// C[M][N] (f32) = A[M][K] (bf16, row-major) * Bt[N][K] (bf16, pre-transposed)
// 256 threads = 8 waves; block tile 128x64; wave tile 32x32; K-step 32.
// Double-buffered LDS staging fed by async global->LDS b128 copies.
// Pipeline tail is peeled so the steady-state loop has NO conditionals.
__global__ void __launch_bounds__(256)
wmma_gemm_kernel(const __bf16* __restrict__ A, const __bf16* __restrict__ Bt,
                 float* __restrict__ C, int M, int N, int K) {
    __shared__ __bf16 sA[2][BM * ASTR];
    __shared__ __bf16 sB[2][BN * ASTR];

    const int tid  = threadIdx.x;
    const int lane = tid & 31;
    const int wave = tid >> 5;
    const int m_blk = blockIdx.x * BM;
    const int n_blk = blockIdx.y * BN;
    const int wm = (wave & 3) * 32;        // wave tile origin inside block
    const int wn = (wave >> 2) * 32;
    const int r16  = lane & 15;
    const int half = lane >> 4;

    // stage-copy geometry: each thread moves 16B (8 bf16) chunks
    const int srow = tid >> 2;             // 0..63
    const int schk = (tid & 3) * 8;        // elem offset 0,8,16,24
    const int ksteps = K / BKK;

    f32x8 c00 = {}, c01 = {}, c10 = {}, c11 = {};

    auto issue_stage = [&](int ks, int buf) {
        const size_t gk = (size_t)ks * BKK;
        cp16_g2l(A  + (size_t)(m_blk + srow)      * K + gk + schk,
                 &sA[buf][srow * ASTR + schk]);
        cp16_g2l(A  + (size_t)(m_blk + 64 + srow) * K + gk + schk,
                 &sA[buf][(64 + srow) * ASTR + schk]);
        cp16_g2l(Bt + (size_t)(n_blk + srow)      * K + gk + schk,
                 &sB[buf][srow * ASTR + schk]);
    };

    auto compute_stage = [&](int buf) {
        const __bf16* pa0 = &sA[buf][(wm + r16)      * ASTR];
        const __bf16* pa1 = &sA[buf][(wm + 16 + r16) * ASTR];
        const __bf16* pb0 = &sB[buf][(wn + r16)      * ASTR];
        const __bf16* pb1 = &sB[buf][(wn + 16 + r16) * ASTR];
        // A frag: two 8-elem runs (K = 8h..8h+7 and 16+8h..16+8h+7)
        bf16x16 a0 = frag16(pa0 + 8 * half, pa0 + 16 + 8 * half);
        bf16x16 a1 = frag16(pa1 + 8 * half, pa1 + 16 + 8 * half);
        // B frag: two 8-elem runs (K = 8h..8h+7 and 16+8h..16+8h+7)
        bf16x16 b0 = frag16(pb0 + 8 * half, pb0 + 16 + 8 * half);
        bf16x16 b1 = frag16(pb1 + 8 * half, pb1 + 16 + 8 * half);
        c00 = __builtin_amdgcn_wmma_f32_16x16x32_bf16(false, a0, false, b0, (short)0, c00, false, false);
        c01 = __builtin_amdgcn_wmma_f32_16x16x32_bf16(false, a0, false, b1, (short)0, c01, false, false);
        c10 = __builtin_amdgcn_wmma_f32_16x16x32_bf16(false, a1, false, b0, (short)0, c10, false, false);
        c11 = __builtin_amdgcn_wmma_f32_16x16x32_bf16(false, a1, false, b1, (short)0, c11, false, false);
    };

    issue_stage(0, 0);
    if (ksteps > 1) issue_stage(1, 1);

    // steady state: unconditional wait / compute / refill
    for (int ks = 0; ks < ksteps - 2; ++ks) {
        wait_async_le3();                 // stage ks landed in LDS
        __syncthreads();
        compute_stage(ks & 1);
        __syncthreads();                  // all waves done reading buf
        issue_stage(ks + 2, ks & 1);
    }
    // stage ksteps-2 (no refill left beyond ksteps-1)
    if (ksteps >= 2) {
        wait_async_le3();
        __syncthreads();
        compute_stage(ksteps & 1);        // (ksteps-2) & 1 == ksteps & 1
        __syncthreads();
    }
    // final stage
    wait_async_zero();
    __syncthreads();
    compute_stage((ksteps - 1) & 1);

    // C layout: lanes 0-15 -> M=j, lanes 16-31 -> M=j+8; N = lane&15
    const int cm = half * 8;
    const int cn = r16;
    const int gm0 = m_blk + wm, gn0 = n_blk + wn;
    #pragma unroll
    for (int j = 0; j < 8; ++j) {
        C[(size_t)(gm0 + cm + j)      * N + (gn0 + cn)]      = c00[j];
        C[(size_t)(gm0 + cm + j)      * N + (gn0 + 16 + cn)] = c01[j];
        C[(size_t)(gm0 + 16 + cm + j) * N + (gn0 + cn)]      = c10[j];
        C[(size_t)(gm0 + 16 + cm + j) * N + (gn0 + 16 + cn)] = c11[j];
    }
}

// -------------------------- bias + causal depthwise conv(3) + SiLU + gate sig
__global__ void conv_silu_gate_kernel(const float* __restrict__ proj,
                                      const float* __restrict__ b_in,
                                      const float* __restrict__ conv_w,
                                      const float* __restrict__ conv_b,
                                      float* __restrict__ vals_f,
                                      __bf16* __restrict__ vals_bf,
                                      float* __restrict__ sgate) {
    const size_t total = (size_t)ROWS * INNER;
    for (size_t e = (size_t)blockIdx.x * blockDim.x + threadIdx.x; e < total;
         e += (size_t)gridDim.x * blockDim.x) {
        const int c = (int)(e % INNER);
        const size_t bs = e / INNER;
        const int s = (int)(bs % SEQ);
        const size_t rb = bs * NPROJ;
        const float bc = b_in[c];
        const float v0  = proj[rb + c] + bc;
        const float vm1 = (s >= 1) ? proj[rb - (size_t)NPROJ + c] + bc : 0.f;
        const float vm2 = (s >= 2) ? proj[rb - 2 * (size_t)NPROJ + c] + bc : 0.f;
        float y = conv_w[c * 3 + 0] * vm2 + conv_w[c * 3 + 1] * vm1 +
                  conv_w[c * 3 + 2] * v0 + conv_b[c];
        y = y * sigmf(y);                      // SiLU
        vals_f[e]  = y;
        vals_bf[e] = f2bf(y);
        sgate[e] = sigmf(proj[rb + INNER + c] + b_in[INNER + c]);
    }
}

// --------------------------------------------- deltas = sigmoid(. + b_delta)
__global__ void sigmoid_bias_kernel(float* __restrict__ dl,
                                    const float* __restrict__ b_delta) {
    const size_t total = (size_t)ROWS * INNER;
    for (size_t e = (size_t)blockIdx.x * blockDim.x + threadIdx.x; e < total;
         e += (size_t)gridDim.x * blockDim.x) {
        dl[e] = sigmf(dl[e] + b_delta[e % INNER]);
    }
}

// ------------------------------------------- fwd + bwd linear recurrence scan
__global__ void scan_kernel(const float* __restrict__ vals_f,
                            const float* __restrict__ deltas,
                            const float* __restrict__ sgate,
                            float* __restrict__ fwdbuf,
                            __bf16* __restrict__ fused_bf) {
    const int t = blockIdx.x * blockDim.x + threadIdx.x;
    if (t >= BATCH * INNER) return;
    const int b = t / INNER, c = t % INNER;
    const size_t base = (size_t)b * SEQ * INNER + c;
    float y = 0.f;
    for (int s = 0; s < SEQ; ++s) {
        const size_t i = base + (size_t)s * INNER;
        const float d = deltas[i];
        y = d * y + (1.f - d) * vals_f[i];
        fwdbuf[i] = y;
    }
    y = 0.f;
    for (int s = SEQ - 1; s >= 0; --s) {
        const size_t i = base + (size_t)s * INNER;
        const float d = deltas[i];
        y = d * y + (1.f - d) * vals_f[i];
        fused_bf[i] = f2bf((fwdbuf[i] + y) * sgate[i]);
    }
}

// --------------------------------------------------- out = raw + b_out + x
__global__ void add_out_kernel(const float* __restrict__ outraw,
                               const float* __restrict__ b_out,
                               const float* __restrict__ x,
                               float* __restrict__ out) {
    const size_t total = (size_t)ROWS * D_MODEL;
    for (size_t e = (size_t)blockIdx.x * blockDim.x + threadIdx.x; e < total;
         e += (size_t)gridDim.x * blockDim.x) {
        out[e] = outraw[e] + b_out[e % D_MODEL] + x[e];
    }
}

extern "C" void kernel_launch(void* const* d_in, const int* in_sizes, int n_in,
                              void* d_out, int out_size, void* d_ws, size_t ws_size,
                              hipStream_t stream) {
    const float* x       = (const float*)d_in[0];
    const float* ln_g    = (const float*)d_in[1];
    const float* ln_b    = (const float*)d_in[2];
    const float* W_in    = (const float*)d_in[3];
    const float* b_in    = (const float*)d_in[4];
    const float* conv_w  = (const float*)d_in[5];
    const float* conv_b  = (const float*)d_in[6];
    const float* W_delta = (const float*)d_in[7];
    const float* b_delta = (const float*)d_in[8];
    const float* W_out   = (const float*)d_in[9];
    const float* b_out   = (const float*)d_in[10];
    float* out = (float*)d_out;

    char* p = (char*)d_ws;
    auto alloc = [&](size_t bytes) -> void* {
        void* r = (void*)p;
        p += (bytes + 255) & ~(size_t)255;
        return r;
    };
    __bf16* h_bf     = (__bf16*)alloc((size_t)ROWS * D_MODEL * 2);
    __bf16* Wi_t     = (__bf16*)alloc((size_t)NPROJ * D_MODEL * 2);
    __bf16* Wd_t     = (__bf16*)alloc((size_t)INNER * INNER * 2);
    __bf16* Wo_t     = (__bf16*)alloc((size_t)D_MODEL * INNER * 2);
    float*  proj     = (float*) alloc((size_t)ROWS * NPROJ * 4);
    float*  vals_f   = (float*) alloc((size_t)ROWS * INNER * 4);
    __bf16* vals_bf  = (__bf16*)alloc((size_t)ROWS * INNER * 2);
    float*  sgate    = (float*) alloc((size_t)ROWS * INNER * 4);
    float*  dlog     = (float*) alloc((size_t)ROWS * INNER * 4);
    float*  fwdbuf   = (float*) alloc((size_t)ROWS * INNER * 4);
    __bf16* fused_bf = (__bf16*)alloc((size_t)ROWS * INNER * 2);
    float*  outraw   = (float*) alloc((size_t)ROWS * D_MODEL * 4);

    // 1. LayerNorm -> bf16 activations
    ln_kernel<<<ROWS, 256, 0, stream>>>(x, ln_g, ln_b, h_bf);

    // 2. Weight convert + transpose to bf16 [N][K]
    cvt_t_kernel<<<2048, 256, 0, stream>>>(W_in,    Wi_t, D_MODEL, NPROJ);
    cvt_t_kernel<<<2048, 256, 0, stream>>>(W_delta, Wd_t, INNER,   INNER);
    cvt_t_kernel<<<1024, 256, 0, stream>>>(W_out,   Wo_t, INNER,   D_MODEL);

    // 3. GEMM1: proj = h @ W_in   (8192x1024 @ 1024x4096)
    {
        dim3 g(ROWS / BM, NPROJ / BN);
        wmma_gemm_kernel<<<g, 256, 0, stream>>>(h_bf, Wi_t, proj, ROWS, NPROJ, D_MODEL);
    }

    // 4. bias + causal conv + SiLU + sigmoid(gate)
    conv_silu_gate_kernel<<<4096, 256, 0, stream>>>(proj, b_in, conv_w, conv_b,
                                                    vals_f, vals_bf, sgate);

    // 5. GEMM2: dlog = values @ W_delta (8192x2048 @ 2048x2048)
    {
        dim3 g(ROWS / BM, INNER / BN);
        wmma_gemm_kernel<<<g, 256, 0, stream>>>(vals_bf, Wd_t, dlog, ROWS, INNER, INNER);
    }

    // 6. deltas = sigmoid(dlog + b_delta), in place
    sigmoid_bias_kernel<<<4096, 256, 0, stream>>>(dlog, b_delta);

    // 7. fwd/bwd selective scan + gating -> fused bf16
    scan_kernel<<<(BATCH * INNER) / 256, 256, 0, stream>>>(vals_f, dlog, sgate,
                                                           fwdbuf, fused_bf);

    // 8. GEMM3: outraw = fused @ W_out (8192x2048 @ 2048x1024)
    {
        dim3 g(ROWS / BM, D_MODEL / BN);
        wmma_gemm_kernel<<<g, 256, 0, stream>>>(fused_bf, Wo_t, outraw, ROWS, D_MODEL, INNER);
    }

    // 9. residual + bias
    add_out_kernel<<<4096, 256, 0, stream>>>(outraw, b_out, x, out);
}
